// NNHead_83288005804317
// MI455X (gfx1250) — compile-verified, matrix operand order checked by reading, not run on hardware
//
#include <hip/hip_runtime.h>
#include <stdint.h>

typedef __attribute__((ext_vector_type(16))) __bf16 v16bf;
typedef __attribute__((ext_vector_type(8)))  float  v8f;

#define BM       64
#define BN       128
#define BK       32
#define CMAX     100
#define LDAB     40      /* padded LDS row stride in bf16 elems (80 bytes, 16B-multiple) */
#define NTHREADS 256
#define NCHUNK   2048    /* db rows per block (N split) */

static __device__ __forceinline__ v16bf wmma_frag_load(const __bf16* base, int row, int half) {
    // 16-bit operand layout (wave32): lane = half*16 + row; two contiguous
    // 8-element groups at K = g*16 + half*8 (g = 0,1)  ->  two ds_read_b128.
    const __bf16* p = base + row * LDAB + half * 8;
    union { uint4 q[2]; v16bf v; } u;
    u.q[0] = *(const uint4*)(p);
    u.q[1] = *(const uint4*)(p + 16);
    return u.v;
}

static __device__ __forceinline__ void split4(float4 v, uint2* hi, uint2* lo) {
    union { __bf16 e[4]; uint2 u; } h, l;
    h.e[0] = (__bf16)v.x; l.e[0] = (__bf16)(v.x - (float)h.e[0]);
    h.e[1] = (__bf16)v.y; l.e[1] = (__bf16)(v.y - (float)h.e[1]);
    h.e[2] = (__bf16)v.z; l.e[2] = (__bf16)(v.z - (float)h.e[2]);
    h.e[3] = (__bf16)v.w; l.e[3] = (__bf16)(v.w - (float)h.e[3]);
    *hi = h.u; *lo = l.u;
}

// -------- row squared-norms: one wave32 per row -----------------------------
__global__ __launch_bounds__(256)
void rownorm_kernel(const float* __restrict__ src, float* __restrict__ dst,
                    int rows, int D) {
    int row  = blockIdx.x * 8 + (threadIdx.x >> 5);
    int lane = threadIdx.x & 31;
    if (row >= rows) return;
    const float* p = src + (size_t)row * D;
    float s = 0.f;
    for (int i = lane; i < D; i += 32) { float v = p[i]; s += v * v; }
    #pragma unroll
    for (int off = 16; off > 0; off >>= 1) s += __shfl_down(s, off, 32);
    if (lane == 0) dst[row] = s;
}

// -------- init output to +inf bit pattern -----------------------------------
__global__ __launch_bounds__(256)
void init_bits_kernel(unsigned int* o, int n) {
    int i = blockIdx.x * 256 + threadIdx.x;
    if (i < n) o[i] = 0x7F800000u;
}

// -------- out = -bits_as_float ----------------------------------------------
__global__ __launch_bounds__(256)
void finalize_kernel(float* o, int n) {
    int i = blockIdx.x * 256 + threadIdx.x;
    if (i < n) {
        unsigned int b = ((const unsigned int*)o)[i];
        o[i] = -__uint_as_float(b);
    }
}

// -------- fused split-bf16 GEMM + per-class segmented min -------------------
__global__ __launch_bounds__(NTHREADS)
void nn_gemm_kernel(const float* __restrict__ x, const float* __restrict__ db,
                    const int* __restrict__ labels,
                    const float* __restrict__ x2, const float* __restrict__ db2,
                    unsigned int* __restrict__ outBits,
                    int Q, int N, int D, int C, int nPerChunk)
{
    __shared__ __align__(16) __bf16 sAhi[BM * LDAB];
    __shared__ __align__(16) __bf16 sAlo[BM * LDAB];
    __shared__ __align__(16) __bf16 sBhi[BN * LDAB];
    __shared__ __align__(16) __bf16 sBlo[BN * LDAB];
    __shared__ int   sCls[BM * CMAX];
    __shared__ float sX2[BM];

    const int t    = threadIdx.x;
    const int lane = t & 31;
    const int wave = t >> 5;
    const int wr   = wave >> 2;    // 0..1 : 32-row band
    const int wc   = wave & 3;     // 0..3 : 32-col band
    const int half = lane >> 4;
    const int l15  = lane & 15;

    const int qBase      = blockIdx.x * BM;
    const int nChunkBase = blockIdx.y * nPerChunk;

    for (int i = t; i < BM * CMAX; i += NTHREADS) sCls[i] = 0x7F800000;
    for (int i = t; i < BM;        i += NTHREADS) sX2[i]  = x2[qBase + i];
    __syncthreads();

    const int kSteps = D / BK;
    const int nTiles = nPerChunk / BN;

    for (int nt = 0; nt < nTiles; ++nt) {
        const int nBase = nChunkBase + nt * BN;
        v8f acc[2][2];
        acc[0][0] = (v8f){}; acc[0][1] = (v8f){};
        acc[1][0] = (v8f){}; acc[1][1] = (v8f){};

        for (int ks = 0; ks < kSteps; ++ks) {
            const int k0 = ks * BK;
            __syncthreads();

            // ---- stage A tile: 64x32 fp32 -> bf16 hi/lo (512 float4, 2/thr)
            #pragma unroll
            for (int j = 0; j < 2; ++j) {
                int c   = t + j * NTHREADS;
                int row = c >> 3;
                int kq  = (c & 7) * 4;
                const float* gp = x + (size_t)(qBase + row) * D + k0 + kq;
                float4 v = *(const float4*)gp;
                if (ks + 1 < kSteps) __builtin_prefetch(gp + BK, 0, 1);
                uint2 h, l; split4(v, &h, &l);
                *(uint2*)(sAhi + row * LDAB + kq) = h;
                *(uint2*)(sAlo + row * LDAB + kq) = l;
            }
            // ---- stage B tile: 128x32 fp32 -> bf16 hi/lo (1024 float4, 4/thr)
            #pragma unroll
            for (int j = 0; j < 4; ++j) {
                int c   = t + j * NTHREADS;
                int row = c >> 3;
                int kq  = (c & 7) * 4;
                const float* gp = db + (size_t)(nBase + row) * D + k0 + kq;
                float4 v = *(const float4*)gp;
                if (ks + 1 < kSteps) __builtin_prefetch(gp + BK, 0, 1);
                uint2 h, l; split4(v, &h, &l);
                *(uint2*)(sBhi + row * LDAB + kq) = h;
                *(uint2*)(sBlo + row * LDAB + kq) = l;
            }
            __syncthreads();

            // ---- fragments
            v16bf aH[2], aL[2], bH[2], bL[2];
            #pragma unroll
            for (int mi = 0; mi < 2; ++mi) {
                int row = wr * 32 + mi * 16 + l15;
                aH[mi] = wmma_frag_load(sAhi, row, half);
                aL[mi] = wmma_frag_load(sAlo, row, half);
            }
            #pragma unroll
            for (int ni = 0; ni < 2; ++ni) {
                int row = wc * 32 + ni * 16 + l15;
                bH[ni] = wmma_frag_load(sBhi, row, half);
                bL[ni] = wmma_frag_load(sBlo, row, half);
            }

            // ---- 3-pass split-precision WMMA: hi*hi + hi*lo + lo*hi
            #pragma unroll
            for (int mi = 0; mi < 2; ++mi) {
                #pragma unroll
                for (int ni = 0; ni < 2; ++ni) {
                    acc[mi][ni] = __builtin_amdgcn_wmma_f32_16x16x32_bf16(
                        false, aH[mi], false, bH[ni], (short)0, acc[mi][ni], false, false);
                    acc[mi][ni] = __builtin_amdgcn_wmma_f32_16x16x32_bf16(
                        false, aH[mi], false, bL[ni], (short)0, acc[mi][ni], false, false);
                    acc[mi][ni] = __builtin_amdgcn_wmma_f32_16x16x32_bf16(
                        false, aL[mi], false, bH[ni], (short)0, acc[mi][ni], false, false);
                }
            }
        }

        // ---- epilogue: dist + per-class LDS min (dist >= 0 -> int-bit min ok)
        #pragma unroll
        for (int ni = 0; ni < 2; ++ni) {
            int n     = nBase + wc * 32 + ni * 16 + l15;
            float dbn = db2[n];
            int   cls = labels[n];
            cls = (cls < 0) ? 0 : (cls >= C ? C - 1 : cls);
            #pragma unroll
            for (int mi = 0; mi < 2; ++mi) {
                v8f a = acc[mi][ni];
                #pragma unroll
                for (int r = 0; r < 8; ++r) {
                    int   mloc = wr * 32 + mi * 16 + r + 8 * half;
                    float d2v  = fmaf(-2.0f, a[r], sX2[mloc] + dbn);
                    float dist = sqrtf(fmaxf(d2v, 0.0f));
                    atomicMin(&sCls[mloc * CMAX + cls], __float_as_int(dist));
                }
            }
        }
    }

    __syncthreads();
    // ---- flush block-local class minima to global (uint-bit min)
    for (int i = t; i < BM * C; i += NTHREADS) {
        int q = i / C, c = i % C;
        int v = sCls[q * CMAX + c];
        if (v != 0x7F800000)
            atomicMin(outBits + (size_t)(qBase + q) * C + c, (unsigned int)v);
    }
}

extern "C" void kernel_launch(void* const* d_in, const int* in_sizes, int n_in,
                              void* d_out, int out_size, void* d_ws, size_t ws_size,
                              hipStream_t stream) {
    const float* x      = (const float*)d_in[0];
    const float* db     = (const float*)d_in[1];
    const int*   labels = (const int*)d_in[2];

    const int N = in_sizes[2];
    const int D = (int)((long long)in_sizes[1] / N);   // 512
    const int Q = (int)((long long)in_sizes[0] / D);   // 2048
    const int C = out_size / Q;                        // 100

    float* ws  = (float*)d_ws;
    float* x2  = ws;
    float* db2 = ws + Q;
    unsigned int* outBits = (unsigned int*)d_out;

    // 1) init output to +inf bits
    init_bits_kernel<<<(Q * C + 255) / 256, 256, 0, stream>>>(outBits, Q * C);

    // 2) exact fp32 row norms
    rownorm_kernel<<<(Q + 7) / 8, 256, 0, stream>>>(x,  x2,  Q, D);
    rownorm_kernel<<<(N + 7) / 8, 256, 0, stream>>>(db, db2, N, D);

    // 3) fused split-bf16 WMMA GEMM + segmented class-min
    int nPerChunk = (N % NCHUNK == 0) ? NCHUNK : N;    // harness: 32768 -> 2048
    dim3 grid(Q / BM, N / nPerChunk);
    nn_gemm_kernel<<<grid, NTHREADS, 0, stream>>>(x, db, labels, x2, db2,
                                                  outBits, Q, N, D, C, nPerChunk);

    // 4) out = -min_dist
    finalize_kernel<<<(Q * C + 255) / 256, 256, 0, stream>>>((float*)d_out, Q * C);
}
